// HeteroLSTMConv_59854664237721
// MI455X (gfx1250) — matrix operand order repaired and checked by reading, not compiled, not run on hardware
//
#include <hip/hip_runtime.h>
#include <math.h>

// ---------------------------------------------------------------------------
// Types
// ---------------------------------------------------------------------------
typedef __bf16 bf16;
typedef __attribute__((ext_vector_type(16))) __bf16 v16bf;
typedef __attribute__((ext_vector_type(8)))  __bf16 v8bf;
typedef __attribute__((ext_vector_type(8)))  float  v8f;
typedef unsigned short __attribute__((ext_vector_type(8))) us8;

#if defined(__gfx1250__) && __has_builtin(__builtin_amdgcn_global_load_async_to_lds_b128) && \
    __has_builtin(__builtin_amdgcn_s_wait_asynccnt)
#define USE_ASYNC 1
#else
#define USE_ASYNC 0
#endif

// exact pointee type of the async-copy builtin params: V4i with AS1 / AS3
typedef int i32x4 __attribute__((vector_size(4 * sizeof(int))));
typedef __attribute__((address_space(1))) i32x4 gi32x4;
typedef __attribute__((address_space(3))) i32x4 li32x4;

static __device__ __forceinline__ bf16 f2bf(float f) {
    unsigned u = __builtin_bit_cast(unsigned, f);
    unsigned short s = (unsigned short)((u + 0x7FFFu + ((u >> 16) & 1u)) >> 16);
    return __builtin_bit_cast(bf16, s);
}
static __device__ __forceinline__ float sigm(float x) { return 1.f / (1.f + __expf(-x)); }
static __device__ __forceinline__ v16bf cat8(v8bf lo, v8bf hi) {
    return __builtin_shufflevector(lo, hi, 0, 1, 2, 3, 4, 5, 6, 7, 8, 9, 10, 11, 12, 13, 14, 15);
}

// ---------------------------------------------------------------------------
// bf16 WMMA GEMM:  C[M,N] = A[M,K]*B[K,N] (+ A2*B2) (+ b1[n]+b2[n])
// A row-major [M,K]; B supplied TRANSPOSED: Bt[N,K] row-major.
// N % 64 == 0, K % 32 == 0. Block = 128 threads (4 waves); tile 128x64:
// wave w owns rows 32w..32w+31 (two 16-row A fragments) x four 16-col tiles.
// Async double-buffered global->LDS staging (GLOBAL_LOAD_ASYNC_TO_LDS_B128 +
// s_wait_asynccnt); sync staging fallback otherwise.
// OOB rows are clamped to M-1 (garbage only reaches C rows >= M, never stored).
// ---------------------------------------------------------------------------
#define TM 128
#define TN 64
#if USE_ASYNC
#define NBUF 2
#else
#define NBUF 1
#endif

__global__ void gemm_bf16_wmma(const bf16* __restrict__ A, const bf16* __restrict__ Bt, int K,
                               const bf16* __restrict__ A2, const bf16* __restrict__ B2t, int K2,
                               const float* __restrict__ bias1, const float* __restrict__ bias2,
                               float* __restrict__ Cout, int M, int N) {
    __shared__ __align__(16) bf16 As[NBUF][TM][32];
    __shared__ __align__(16) bf16 Bs[NBUF][TN][32];
    const int tid  = threadIdx.x;
    const int lane = tid & 31;
    const int wave = tid >> 5;
    const int m0 = blockIdx.x * TM;
    const int n0 = blockIdx.y * TN;
    const int lrow = lane & 15;
    const int half = lane >> 4;
    const int colseg = (tid & 3) * 8;   // element offset of this thread's 8-elem chunk
    const int rbase  = tid >> 2;        // row base (+32 per step)

    const int S1 = K / 32;
    const int S2 = A2 ? (K2 / 32) : 0;
    const int S  = S1 + S2;

    v8f acc[2][4] = {};

    auto tileOf = [&](int s, const bf16*& Ap, const bf16*& Bp, int& k0, int& Kp) {
        if (s < S1) { Ap = A;  Bp = Bt;  k0 = s * 32;        Kp = K;  }
        else        { Ap = A2; Bp = B2t; k0 = (s - S1) * 32; Kp = K2; }
    };

    auto computeTile = [&](int buf) {
        v16bf afrag[2];
#pragma unroll
        for (int mi = 0; mi < 2; ++mi) {
            int m = wave * 32 + mi * 16 + lrow;
            afrag[mi] = cat8(*(const v8bf*)&As[buf][m][8 * half],
                             *(const v8bf*)&As[buf][m][16 + 8 * half]);
        }
#pragma unroll
        for (int t = 0; t < 4; ++t) {
            int n = t * 16 + lrow;
            v16bf bfrag = cat8(*(const v8bf*)&Bs[buf][n][8 * half],
                               *(const v8bf*)&Bs[buf][n][16 + 8 * half]);
            acc[0][t] = __builtin_amdgcn_wmma_f32_16x16x32_bf16(
                false, afrag[0], false, bfrag, (short)0, acc[0][t], false, false);
            acc[1][t] = __builtin_amdgcn_wmma_f32_16x16x32_bf16(
                false, afrag[1], false, bfrag, (short)0, acc[1][t], false, false);
        }
    };

#if USE_ASYNC
    auto issueTile = [&](int s, int buf) {
        const bf16 *Ap, *Bp; int k0, Kp; tileOf(s, Ap, Bp, k0, Kp);
#pragma unroll
        for (int it = 0; it < 4; ++it) {               // A: 128 rows x 32 k
            int r = rbase + it * 32;
            int gm = (m0 + r < M) ? (m0 + r) : (M - 1);
            const unsigned short* gp = (const unsigned short*)Ap + (size_t)gm * Kp + k0 + colseg;
            __builtin_amdgcn_global_load_async_to_lds_b128(
                (gi32x4*)(void*)gp, (li32x4*)(void*)&As[buf][r][colseg], 0, 0);
        }
#pragma unroll
        for (int it = 0; it < 2; ++it) {               // B: 64 n x 32 k
            int r = rbase + it * 32;
            const unsigned short* gp = (const unsigned short*)Bp + (size_t)(n0 + r) * Kp + k0 + colseg;
            __builtin_amdgcn_global_load_async_to_lds_b128(
                (gi32x4*)(void*)gp, (li32x4*)(void*)&Bs[buf][r][colseg], 0, 0);
        }
    };

    issueTile(0, 0);
    for (int s = 0; s < S; ++s) {
        int buf = s & 1;
        if (s + 1 < S) {
            issueTile(s + 1, buf ^ 1);
            __builtin_amdgcn_s_wait_asynccnt(6);   // tile s resident (in-order completion)
        } else {
            __builtin_amdgcn_s_wait_asynccnt(0);
        }
        __syncthreads();
        computeTile(buf);
        __syncthreads();                           // all waves done reading before reuse
    }
#else
    for (int s = 0; s < S; ++s) {
        const bf16 *Ap, *Bp; int k0, Kp; tileOf(s, Ap, Bp, k0, Kp);
        __syncthreads();
#pragma unroll
        for (int it = 0; it < 4; ++it) {
            int r = rbase + it * 32;
            int gm = (m0 + r < M) ? (m0 + r) : (M - 1);
            *(us8*)&As[0][r][colseg] =
                *(const us8*)((const unsigned short*)Ap + (size_t)gm * Kp + k0 + colseg);
        }
#pragma unroll
        for (int it = 0; it < 2; ++it) {
            int r = rbase + it * 32;
            *(us8*)&Bs[0][r][colseg] =
                *(const us8*)((const unsigned short*)Bp + (size_t)(n0 + r) * Kp + k0 + colseg);
        }
        if (s + 1 < S) {
            const bf16 *Apn, *Bpn; int k0n, Kpn; tileOf(s + 1, Apn, Bpn, k0n, Kpn);
            __builtin_prefetch((const void*)((const unsigned short*)Bpn +
                               (size_t)(n0 + lane) * Kpn + k0n), 0, 1);
        }
        __syncthreads();
        computeTile(0);
    }
#endif

    // store: C/D layout -> VGPR r holds row m = r + 8*half, col n = 16t + (lane%16)
#pragma unroll
    for (int t = 0; t < 4; ++t) {
        int n = n0 + t * 16 + lrow;
        float bv = 0.f;
        if (bias1) bv += bias1[n];
        if (bias2) bv += bias2[n];
#pragma unroll
        for (int mi = 0; mi < 2; ++mi) {
#pragma unroll
            for (int r = 0; r < 8; ++r) {
                int m = m0 + wave * 32 + mi * 16 + r + 8 * half;
                if (m < M) Cout[(size_t)m * N + n] = acc[mi][t][r] + bv;
            }
        }
    }
}

// ---------------------------------------------------------------------------
// fp32 -> bf16 converts (plain and transposed)
// ---------------------------------------------------------------------------
__global__ void cvt_bf16_kernel(const float* __restrict__ in, bf16* __restrict__ out, size_t n) {
    size_t i = (size_t)blockIdx.x * blockDim.x + threadIdx.x;
    if (i < n) out[i] = f2bf(in[i]);
}
__global__ void cvt_bf16_T_kernel(const float* __restrict__ in, bf16* __restrict__ out, int R, int Cc) {
    size_t i = (size_t)blockIdx.x * blockDim.x + threadIdx.x;
    if (i >= (size_t)R * Cc) return;
    int r = (int)(i / Cc), c = (int)(i % Cc);
    out[(size_t)c * R + r] = f2bf(in[i]);     // out[Cc,R] = in[R,Cc]^T
}
__global__ void zero_kernel(float* __restrict__ p, size_t n) {
    size_t i = (size_t)blockIdx.x * blockDim.x + threadIdx.x;
    for (; i < n; i += (size_t)gridDim.x * blockDim.x) p[i] = 0.f;
}

// ---------------------------------------------------------------------------
// Edge messages: one wave per edge. g = sx[src] viewed [8,32]; alpha_h = <g_h, a_h>;
// scatter-add g*alpha into agg[dst]. Wave32 shuffle reduction per head.
// ---------------------------------------------------------------------------
__global__ void edge_msg_kernel(const float* __restrict__ sx, const float* __restrict__ a,
                                const int* __restrict__ src, const int* __restrict__ dst,
                                float* __restrict__ agg, int nE) {
    int wave = (int)((blockIdx.x * (size_t)blockDim.x + threadIdx.x) >> 5);
    int lane = threadIdx.x & 31;
    if (wave >= nE) return;
    size_t s = (size_t)src[wave] * 256;
    size_t d = (size_t)dst[wave] * 256;
#pragma unroll
    for (int h = 0; h < 8; ++h) {
        float val = sx[s + h * 32 + lane];
        float p = val * a[h * 32 + lane];
#pragma unroll
        for (int off = 16; off; off >>= 1) p += __shfl_xor(p, off, 32);
        atomicAdd(&agg[d + h * 32 + lane], val * p);
    }
}

// ---------------------------------------------------------------------------
// LSTM pointwise: gates[M,1024] (i|f|g|o); c0 scaled by cscale.
// h = sig(o)*tanh(sig(f)*c + sig(i)*tanh(g)); optional relu; optional bf16 copy.
// ---------------------------------------------------------------------------
__global__ void lstm_act_kernel(const float* __restrict__ gates, const float* __restrict__ c0,
                                float cscale, float* __restrict__ hout, bf16* __restrict__ hbf,
                                int M, int doRelu) {
    size_t idx = (size_t)blockIdx.x * blockDim.x + threadIdx.x;
    size_t total = (size_t)M * 256;
    if (idx >= total) return;
    size_t row = idx >> 8; int col = (int)(idx & 255);
    const float* g = gates + row * 1024;
    float ig = sigm(g[col]);
    float fg = sigm(g[256 + col]);
    float gg = tanhf(g[512 + col]);
    float og = sigm(g[768 + col]);
    float c2 = fg * (c0[idx] * cscale) + ig * gg;
    float h  = og * tanhf(c2);
    if (doRelu) h = fmaxf(h, 0.f);
    hout[idx] = h;
    if (hbf) hbf[idx] = f2bf(h);
}

// column sums of tanh(x) over M rows (row stride 256); atomic accumulate
__global__ void tanh_colsum_kernel(const float* __restrict__ x, float* __restrict__ colsum, int M) {
    int col = threadIdx.x;                 // 256 threads
    float s = 0.f;
    for (int r = blockIdx.x; r < M; r += gridDim.x) s += tanhf(x[(size_t)r * 256 + col]);
    atomicAdd(&colsum[col], s);
}

// scores over 2 types -> softmax attn[2]
__global__ void scores_kernel(const float* __restrict__ colsum, const float* __restrict__ q,
                              float invM, float* __restrict__ attn) {
    __shared__ float s0s[256], s1s[256];
    int t = threadIdx.x;
    float qv = q[t];
    s0s[t] = qv * colsum[t] * invM;
    s1s[t] = qv * colsum[256 + t] * invM;
    __syncthreads();
    for (int off = 128; off; off >>= 1) {
        if (t < off) { s0s[t] += s0s[t + off]; s1s[t] += s1s[t + off]; }
        __syncthreads();
    }
    if (t == 0) {
        float a = s0s[0], b = s1s[0];
        float m = fmaxf(a, b);
        float e0 = __expf(a - m), e1 = __expf(b - m);
        attn[0] = e0 / (e0 + e1);
        attn[1] = e1 / (e0 + e1);
    }
}

__global__ void combine_kernel(const float* __restrict__ hv, const float* __restrict__ hn,
                               const float* __restrict__ attn, float* __restrict__ out, size_t n) {
    size_t i = (size_t)blockIdx.x * blockDim.x + threadIdx.x;
    if (i < n) out[i] = fmaxf(attn[0] * hv[i] + attn[1] * hn[i], 0.f);
}

// ---------------------------------------------------------------------------
// Host-side launch
// ---------------------------------------------------------------------------
static inline size_t cdiv(size_t a, size_t b) { return (a + b - 1) / b; }

extern "C" void kernel_launch(void* const* d_in, const int* in_sizes, int n_in,
                              void* d_out, int out_size, void* d_ws, size_t ws_size,
                              hipStream_t stream) {
    (void)in_sizes; (void)n_in; (void)out_size; (void)ws_size;
    const int NU = 100000, NP = 50000, C = 256, E = 400000, KIN = 128, G = 1024;

    const float* x_user = (const float*)d_in[0];
    const float* x_poi  = (const float*)d_in[1];
    // tag bases: v=2, n=9, r=16 -> Wsrc, Wtgt, a, Wih, Whh, bih, bhh
    const int TB[3] = {2, 9, 16};
    const float* klinb_poi = (const float*)d_in[27];
    const float* q_poi     = (const float*)d_in[28];
    const int* e_src[3] = {(const int*)d_in[29], (const int*)d_in[31], (const int*)d_in[33]};
    const int* e_dst[3] = {(const int*)d_in[30], (const int*)d_in[32], (const int*)d_in[34]};

    // workspace bump allocator (256B aligned)
    char* ws = (char*)d_ws;
    size_t off = 0;
    auto alloc = [&](size_t bytes) -> void* {
        off = (off + 255) & ~(size_t)255;
        void* p = ws + off;
        off += bytes;
        return p;
    };
    // fp32 buffers
    float* sx[3];  sx[0] = (float*)alloc((size_t)NU * C * 4);  // v: src = user
                   sx[1] = (float*)alloc((size_t)NP * C * 4);
                   sx[2] = (float*)alloc((size_t)NP * C * 4);
    float* tx[3];  tx[0] = (float*)alloc((size_t)NP * C * 4);  // v: tgt = poi
                   tx[1] = (float*)alloc((size_t)NP * C * 4);
                   tx[2] = (float*)alloc((size_t)NU * C * 4);  // r: tgt = user
    float* agg[3]; agg[0] = (float*)alloc((size_t)NP * C * 4);
                   agg[1] = (float*)alloc((size_t)NP * C * 4);
                   agg[2] = (float*)alloc((size_t)NU * C * 4);
    float* gates = (float*)alloc((size_t)NU * G * 4);          // shared scratch (max M)
    float* h_v   = (float*)alloc((size_t)NP * C * 4);
    float* h_n   = (float*)alloc((size_t)NP * C * 4);
    float* colsum = (float*)alloc(512 * 4);
    float* attn   = (float*)alloc(2 * 4);
    // bf16 buffers (all GEMM B operands stored as Bt[N,K] row-major)
    bf16* xu_b = (bf16*)alloc((size_t)NU * KIN * 2);
    bf16* xp_b = (bf16*)alloc((size_t)NP * KIN * 2);
    bf16 *WsrcT_b[3], *WtgtT_b[3], *Wih_b[3], *Whh_b[3];
    for (int t = 0; t < 3; ++t) {
        WsrcT_b[t] = (bf16*)alloc((size_t)C * KIN * 2);  // [N=256][K=128]
        WtgtT_b[t] = (bf16*)alloc((size_t)C * KIN * 2);
        Wih_b[t]   = (bf16*)alloc((size_t)G * C * 2);    // Wih itself = Bt for x@Wih^T
        Whh_b[t]   = (bf16*)alloc((size_t)G * C * 2);
    }
    bf16* klW_b = (bf16*)alloc((size_t)C * C * 2);       // klinW itself = Bt
    bf16* tx_b[3];  tx_b[0] = (bf16*)alloc((size_t)NP * C * 2);
                    tx_b[1] = (bf16*)alloc((size_t)NP * C * 2);
                    tx_b[2] = (bf16*)alloc((size_t)NU * C * 2);
    bf16* agg_b[3]; agg_b[0] = (bf16*)alloc((size_t)NP * C * 2);
                    agg_b[1] = (bf16*)alloc((size_t)NP * C * 2);
                    agg_b[2] = (bf16*)alloc((size_t)NU * C * 2);
    bf16* hv_b = (bf16*)alloc((size_t)NP * C * 2);
    bf16* hn_b = (bf16*)alloc((size_t)NP * C * 2);

    float* out_user = (float*)d_out;
    float* out_poi  = (float*)d_out + (size_t)NU * C;

    auto cvt = [&](const float* in, bf16* outp, size_t n) {
        cvt_bf16_kernel<<<dim3((unsigned)cdiv(n, 256)), 256, 0, stream>>>(in, outp, n);
    };
    auto cvtT = [&](const float* in, bf16* outp, int R, int Cc) {
        cvt_bf16_T_kernel<<<dim3((unsigned)cdiv((size_t)R * Cc, 256)), 256, 0, stream>>>(in, outp, R, Cc);
    };
    auto gemm = [&](const bf16* A, const bf16* Bt, int K, const bf16* A2, const bf16* B2t, int K2,
                    const float* b1, const float* b2, float* Cp, int M, int N) {
        gemm_bf16_wmma<<<dim3((unsigned)cdiv(M, TM), (unsigned)(N / TN)), 128, 0, stream>>>(
            A, Bt, K, A2, B2t, K2, b1, b2, Cp, M, N);
    };

    // 1) convert inputs + weights to bf16.  B operands in Bt[N,K] layout:
    //    x@Wsrc needs Bt = Wsrc^T (transpose); x@Wih^T needs Bt = Wih (plain copy).
    cvt(x_user, xu_b, (size_t)NU * KIN);
    cvt(x_poi,  xp_b, (size_t)NP * KIN);
    for (int t = 0; t < 3; ++t) {
        cvtT((const float*)d_in[TB[t] + 0], WsrcT_b[t], KIN, C);  // [128,256] -> [256,128]
        cvtT((const float*)d_in[TB[t] + 1], WtgtT_b[t], KIN, C);
        cvt((const float*)d_in[TB[t] + 3], Wih_b[t], (size_t)G * C);
        cvt((const float*)d_in[TB[t] + 4], Whh_b[t], (size_t)G * C);
    }
    cvt((const float*)d_in[26], klW_b, (size_t)C * C);

    // 2) projections: sx = x_s @ Wsrc ; tx = x_t @ Wtgt
    gemm(xu_b, WsrcT_b[0], KIN, nullptr, nullptr, 0, nullptr, nullptr, sx[0], NU, C); // v src=user
    gemm(xp_b, WtgtT_b[0], KIN, nullptr, nullptr, 0, nullptr, nullptr, tx[0], NP, C); // v tgt=poi
    gemm(xp_b, WsrcT_b[1], KIN, nullptr, nullptr, 0, nullptr, nullptr, sx[1], NP, C); // n
    gemm(xp_b, WtgtT_b[1], KIN, nullptr, nullptr, 0, nullptr, nullptr, tx[1], NP, C);
    gemm(xp_b, WsrcT_b[2], KIN, nullptr, nullptr, 0, nullptr, nullptr, sx[2], NP, C); // r src=poi
    gemm(xu_b, WtgtT_b[2], KIN, nullptr, nullptr, 0, nullptr, nullptr, tx[2], NU, C); // r tgt=user

    // 3) zero agg, then edge gather/score/scatter-add (wave per edge)
    const int aggM[3] = {NP, NP, NU};
    for (int t = 0; t < 3; ++t)
        zero_kernel<<<dim3(2048), 256, 0, stream>>>(agg[t], (size_t)aggM[t] * C);
    for (int t = 0; t < 3; ++t) {
        const float* a_vec = (const float*)d_in[TB[t] + 2];
        edge_msg_kernel<<<dim3((unsigned)cdiv((size_t)E * 32, 256)), 256, 0, stream>>>(
            sx[t], a_vec, e_src[t], e_dst[t], agg[t], E);
    }

    // 4) bf16 copies of tx / agg for the LSTM dual-GEMM
    const int txM[3] = {NP, NP, NU};
    for (int t = 0; t < 3; ++t) {
        cvt(tx[t],  tx_b[t],  (size_t)txM[t] * C);
        cvt(agg[t], agg_b[t], (size_t)aggM[t] * C);
    }

    // 5) per-type: gates = tx@Wih^T + agg@Whh^T + (bih+bhh), then LSTM pointwise.
    //    meta_poi = agg_v / 2 feeds BOTH v and n; meta_user = agg_r.
    gemm(tx_b[0], Wih_b[0], C, agg_b[0], Whh_b[0], C,
         (const float*)d_in[TB[0] + 5], (const float*)d_in[TB[0] + 6], gates, NP, G);
    lstm_act_kernel<<<dim3((unsigned)cdiv((size_t)NP * C, 256)), 256, 0, stream>>>(
        gates, agg[0], 0.5f, h_v, hv_b, NP, 0);
    gemm(tx_b[1], Wih_b[1], C, agg_b[1], Whh_b[1], C,
         (const float*)d_in[TB[1] + 5], (const float*)d_in[TB[1] + 6], gates, NP, G);
    lstm_act_kernel<<<dim3((unsigned)cdiv((size_t)NP * C, 256)), 256, 0, stream>>>(
        gates, agg[0], 0.5f, h_n, hn_b, NP, 0);
    // r: out_user = relu(h_r) directly (softmax over single type == 1)
    gemm(tx_b[2], Wih_b[2], C, agg_b[2], Whh_b[2], C,
         (const float*)d_in[TB[2] + 5], (const float*)d_in[TB[2] + 6], gates, NU, G);
    lstm_act_kernel<<<dim3((unsigned)cdiv((size_t)NU * C, 256)), 256, 0, stream>>>(
        gates, agg[2], 1.0f, out_user, nullptr, NU, 1);

    // 6) poi semantic attention: score_t = q . mean_n tanh(h_t@klinW^T + b)
    zero_kernel<<<dim3(2), 256, 0, stream>>>(colsum, 512);
    gemm(hv_b, klW_b, C, nullptr, nullptr, 0, klinb_poi, nullptr, gates, NP, C);
    tanh_colsum_kernel<<<dim3(512), 256, 0, stream>>>(gates, colsum, NP);
    gemm(hn_b, klW_b, C, nullptr, nullptr, 0, klinb_poi, nullptr, gates, NP, C);
    tanh_colsum_kernel<<<dim3(512), 256, 0, stream>>>(gates, colsum + 256, NP);
    scores_kernel<<<dim3(1), 256, 0, stream>>>(colsum, q_poi, 1.0f / (float)NP, attn);
    combine_kernel<<<dim3((unsigned)cdiv((size_t)NP * C, 256)), 256, 0, stream>>>(
        h_v, h_n, attn, out_poi, (size_t)NP * C);
}
